// muliheaded_attention_30657476559636
// MI455X (gfx1250) — matrix operand order repaired and checked by reading, not compiled, not run on hardware
//
#include <hip/hip_runtime.h>

#define B_   4
#define S_   2048
#define HID_ 512
#define H_   8
#define DH_  64

typedef __attribute__((ext_vector_type(16))) _Float16 v16h;
typedef __attribute__((ext_vector_type(8)))  _Float16 v8h;
typedef __attribute__((ext_vector_type(8)))  float    v8f;

union AFrag { v16h v; unsigned u[8]; _Float16 h[16]; };

__device__ __forceinline__ v8f wmma_f16(v16h a, v16h b, v8f c) {
  // D = A(16x32 f16) x B(32x16 f16) + C(16x16 f32)
  return __builtin_amdgcn_wmma_f32_16x16x32_f16(
      /*neg_a=*/false, a, /*neg_b=*/false, b,
      /*c_mod=*/(short)0, c, /*reuse_a=*/false, /*reuse_b=*/false);
}

// ---------------------------------------------------------------------------
// Kernel 0: convert Wq|Wk|Wv (each [H,DH,DH] f32) into one f16 buffer.
// ---------------------------------------------------------------------------
__global__ void cvt_weights(const float* __restrict__ Wq,
                            const float* __restrict__ Wk,
                            const float* __restrict__ Wv,
                            _Float16* __restrict__ wf16) {
  int gid = blockIdx.x * blockDim.x + threadIdx.x;
  const int per = H_ * DH_ * DH_;  // 32768
  if (gid >= 3 * per) return;
  const float* src = (gid < per) ? Wq : (gid < 2 * per ? Wk : Wv);
  wf16[gid] = (_Float16)src[gid % per];
}

// ---------------------------------------------------------------------------
// Kernel 1: QKV projection. One wave = 16 rows of s for one (b,h).
//   q,k stored row-major f16 [B,H,S,DH]; v stored transposed f16 [B,H,DH,S].
//   Softmax scale 1/sqrt(64)=0.125 folded into q.
// ---------------------------------------------------------------------------
__global__ void __launch_bounds__(256) qkv_proj(
    const float* __restrict__ x, const _Float16* __restrict__ wf16,
    const float* __restrict__ bq, const float* __restrict__ bk,
    const float* __restrict__ bv,
    _Float16* __restrict__ qb, _Float16* __restrict__ kb,
    _Float16* __restrict__ vt) {
  const int ST = S_ / 128;
  int blk = blockIdx.x;
  int b = blk / (H_ * ST);
  int rem = blk % (H_ * ST);
  int h = rem / ST;
  int st = rem % ST;
  int w = threadIdx.x >> 5, lane = threadIdx.x & 31;
  int lgrp = lane >> 4, l15 = lane & 15;
  int s0 = st * 128 + w * 16;
  int row = s0 + l15;

  // A fragments of x (f32 global -> f16 regs), contiguous k order: element s holds k = 16*lgrp + s (same as B).
  AFrag a[2];
#pragma unroll
  for (int ks = 0; ks < 2; ++ks) {
    const float* xr = x + ((size_t)b * S_ + row) * HID_ + h * DH_ + ks * 32;
#pragma unroll
    for (int v = 0; v < 8; ++v) {
      int kidx = lgrp * 16 + 2 * v;
      float2 f = *(const float2*)(xr + kidx);
      a[ks].h[2 * v]     = (_Float16)f.x;
      a[ks].h[2 * v + 1] = (_Float16)f.y;
    }
  }

  const float* biases[3] = {bq, bk, bv};
  size_t headw = (size_t)h * DH_ * DH_;
  size_t outbase = ((size_t)b * H_ + h) * S_;

#pragma unroll
  for (int mat = 0; mat < 3; ++mat) {
    const _Float16* wmat = wf16 + (size_t)mat * H_ * DH_ * DH_ + headw;
#pragma unroll
    for (int nt = 0; nt < 4; ++nt) {
      int e = nt * 16 + l15;
      // B-fragment: source [e][d] row-major, 16 contiguous halves per lane.
      v16h b0 = *(const v16h*)(wmat + (size_t)e * DH_ + lgrp * 16);
      v16h b1 = *(const v16h*)(wmat + (size_t)e * DH_ + 32 + lgrp * 16);
      v8f c = {};
      c = wmma_f16(a[0].v, b0, c);
      c = wmma_f16(a[1].v, b1, c);
      float bias = biases[mat][h * DH_ + e];
#pragma unroll
      for (int i = 0; i < 8; ++i) c[i] += bias;
      if (mat == 0) {
#pragma unroll
        for (int i = 0; i < 8; ++i) c[i] *= 0.125f;  // 1/sqrt(DH)
      }
      if (mat < 2) {
        _Float16* dst = (mat == 0) ? qb : kb;
#pragma unroll
        for (int i = 0; i < 8; ++i) {
          int s = s0 + lgrp * 8 + i;
          dst[(outbase + s) * DH_ + e] = (_Float16)c[i];
        }
      } else {
        v8h o;
#pragma unroll
        for (int i = 0; i < 8; ++i) o[i] = (_Float16)c[i];
        // coalesced 16B store: vt[b,h,d=e,s0 + lgrp*8 .. +7]
        *(v8h*)(vt + (((size_t)b * H_ + h) * DH_ + e) * S_ + s0 + lgrp * 8) = o;
      }
    }
  }
}

// ---------------------------------------------------------------------------
// Kernel 2: flash attention. One wave = 16 query rows, streams 32-key tiles.
// ---------------------------------------------------------------------------
__global__ void __launch_bounds__(256) flash_attn(
    const _Float16* __restrict__ qb, const _Float16* __restrict__ kb,
    const _Float16* __restrict__ vt, float* __restrict__ out) {
  __shared__ _Float16 plds[8][16 * 32];  // per-wave 16x32 P tile (8 KB)
  const int ST = S_ / 128;
  int blk = blockIdx.x;
  int b = blk / (H_ * ST);
  int rem = blk % (H_ * ST);
  int h = rem / ST;
  int st = rem % ST;
  int w = threadIdx.x >> 5, lane = threadIdx.x & 31;
  int lgrp = lane >> 4, l15 = lane & 15;
  int s0 = st * 128 + w * 16;

  size_t base = ((size_t)b * H_ + h) * S_;
  const _Float16* qrow = qb + (base + s0 + l15) * DH_;
  const _Float16* kbp  = kb + base * DH_;
  const _Float16* vbp  = vt + (((size_t)b * H_ + h) * DH_) * S_;

  // Q A-fragments, loaded once (scale already folded in).
  AFrag a[2];
#pragma unroll
  for (int ks = 0; ks < 2; ++ks)
#pragma unroll
    for (int v = 0; v < 8; ++v) {
      int kidx = ks * 32 + lgrp * 16 + 2 * v;
      a[ks].u[v] = *(const unsigned*)(qrow + kidx);
    }

  float m_[8], l_[8], alpha[8];
  v8f acc[4];
  v8f vzero = {};
#pragma unroll
  for (int i = 0; i < 8; ++i) { m_[i] = -1e30f; l_[i] = 0.0f; }
#pragma unroll
  for (int nt = 0; nt < 4; ++nt) acc[nt] = vzero;

  for (int t0 = 0; t0 < S_; t0 += 32) {
    // K B-fragments: K row-major [t][d] -> 32B contiguous per lane.
    v16h kf[2][2];
#pragma unroll
    for (int nt = 0; nt < 2; ++nt)
#pragma unroll
      for (int ks = 0; ks < 2; ++ks)
        kf[nt][ks] = *(const v16h*)(kbp + (size_t)(t0 + nt * 16 + l15) * DH_ +
                                    ks * 32 + lgrp * 16);

    if (t0 + 32 < S_) {  // uniform branch: prefetch next K/V tiles
      __builtin_prefetch(kbp + (size_t)(t0 + 32 + l15) * DH_, 0, 1);
      __builtin_prefetch(vbp + (size_t)l15 * S_ + t0 + 32, 0, 1);
    }

    // scores: 16 queries x 32 keys, f32 accumulate
    v8f sc[2];
#pragma unroll
    for (int nt = 0; nt < 2; ++nt) {
      v8f c = {};
      c = wmma_f16(a[0].v, kf[nt][0], c);
      c = wmma_f16(a[1].v, kf[nt][1], c);
      sc[nt] = c;
    }

    // online softmax in C/D layout: row M = i + 8*lgrp, key N = l15 (+16)
#pragma unroll
    for (int i = 0; i < 8; ++i) {
      float tm = fmaxf(sc[0][i], sc[1][i]);
#pragma unroll
      for (int msk = 1; msk < 16; msk <<= 1)
        tm = fmaxf(tm, __shfl_xor(tm, msk, 32));
      float mn = fmaxf(m_[i], tm);
      alpha[i] = __expf(m_[i] - mn);
      float p0 = __expf(sc[0][i] - mn);
      float p1 = __expf(sc[1][i] - mn);
      float rs = p0 + p1;
#pragma unroll
      for (int msk = 1; msk < 16; msk <<= 1) rs += __shfl_xor(rs, msk, 32);
      l_[i] = l_[i] * alpha[i] + rs;
      m_[i] = mn;
      sc[0][i] = p0;
      sc[1][i] = p1;
    }
#pragma unroll
    for (int nt = 0; nt < 4; ++nt)
#pragma unroll
      for (int i = 0; i < 8; ++i) acc[nt][i] *= alpha[i];

    // P: C-layout -> LDS -> A-layout (per-wave private region, in-order DS)
#pragma unroll
    for (int nt = 0; nt < 2; ++nt)
#pragma unroll
      for (int i = 0; i < 8; ++i)
        plds[w][(lgrp * 8 + i) * 32 + nt * 16 + l15] = (_Float16)sc[nt][i];
    asm volatile("s_wait_dscnt 0" ::: "memory");

    AFrag pa;
#pragma unroll
    for (int v = 0; v < 8; ++v) {
      int kidx = lgrp * 16 + 2 * v;
      pa.u[v] = *(const unsigned*)&plds[w][l15 * 32 + kidx];
    }
    asm volatile("s_wait_dscnt 0" ::: "memory");

    // PV: B-fragments from transposed V (key dim contiguous per lane)
#pragma unroll
    for (int nt = 0; nt < 4; ++nt) {
      v16h vf = *(const v16h*)(vbp + (size_t)(nt * 16 + l15) * S_ + t0 +
                               lgrp * 16);
      acc[nt] = wmma_f16(pa.v, vf, acc[nt]);
    }
  }

  // epilogue: divide by l, scatter f32 to [B,S,HID]
#pragma unroll
  for (int i = 0; i < 8; ++i) l_[i] = 1.0f / l_[i];
#pragma unroll
  for (int nt = 0; nt < 4; ++nt) {
    int e = nt * 16 + l15;
#pragma unroll
    for (int i = 0; i < 8; ++i) {
      int s = s0 + lgrp * 8 + i;
      out[((size_t)b * S_ + s) * HID_ + h * DH_ + e] = acc[nt][i] * l_[i];
    }
  }
}

// ---------------------------------------------------------------------------
extern "C" void kernel_launch(void* const* d_in, const int* in_sizes, int n_in,
                              void* d_out, int out_size, void* d_ws,
                              size_t ws_size, hipStream_t stream) {
  (void)in_sizes; (void)n_in; (void)out_size; (void)ws_size;
  const float* x  = (const float*)d_in[0];
  const float* Wq = (const float*)d_in[1];
  const float* bq = (const float*)d_in[2];
  const float* Wk = (const float*)d_in[3];
  const float* bk = (const float*)d_in[4];
  const float* Wv = (const float*)d_in[5];
  const float* bv = (const float*)d_in[6];
  float* out = (float*)d_out;

  const size_t WBYTES = (size_t)3 * H_ * DH_ * DH_ * 2;        // 196608
  const size_t QKVB   = (size_t)B_ * H_ * S_ * DH_ * 2;        // 8388608
  char* ws = (char*)d_ws;
  _Float16* wf16 = (_Float16*)ws;
  _Float16* qb   = (_Float16*)(ws + WBYTES);
  _Float16* kb   = (_Float16*)(ws + WBYTES + QKVB);
  _Float16* vt   = (_Float16*)(ws + WBYTES + 2 * QKVB);

  cvt_weights<<<(3 * H_ * DH_ * DH_ + 255) / 256, 256, 0, stream>>>(Wq, Wk, Wv,
                                                                    wf16);
  qkv_proj<<<B_ * H_ * (S_ / 128), 256, 0, stream>>>(x, wf16, bq, bk, bv, qb,
                                                     kb, vt);
  flash_attn<<<B_ * H_ * (S_ / 128), 256, 0, stream>>>(qb, kb, vt, out);
}
